// PostProcessCocoGrounding_87076166959330
// MI455X (gfx1250) — matrix operand order repaired and checked by reading, not compiled, not run on hardware
//
#include <hip/hip_runtime.h>

// ---------------------------------------------------------------------------
// PostProcessCocoGrounding on MI455X (gfx1250):
//   prob = (Q x T sigmoid(logits)) @ (T x C pmap^T)  via v_wmma_f32_16x16x32_f16
//   top-300 per batch via exact 32-bit radix select + deterministic rank sort
//   box convert / gather / scale fused into emission
// ---------------------------------------------------------------------------

typedef __attribute__((ext_vector_type(16))) _Float16 v16h;
typedef __attribute__((ext_vector_type(8)))  float    v8f;

#define B_      256
#define Q_      900
#define T_      256
#define C_      91
#define K_SEL   300
#define NTILE   6      // ceil(91/16)  -> padded C = 96
#define MTILE   57     // ceil(900/16) -> padded Q = 912
#define KSTEPS  8      // 256 / 32

// workspace layout (bytes)
#define WS_BFRAG_OFF 0        // 6*8*32*16 halves = 49152 B
#define WS_PROB_OFF  65536    // 256*900*91 f32   = 83,865,600 B

// fast sigmoid: v_exp_f32 + single v_rcp_f32 (result is truncated to f16 for
// WMMA input, so 1-ulp rcp is ample; avoids the precise div_scale/div_fixup
// sequence that dominated VALU issue in the GEMM loop)
__device__ __forceinline__ float sigmoidf_(float x) {
    return __builtin_amdgcn_rcpf(1.0f + __expf(-x));
}

// ---------------------------------------------------------------------------
// Kernel 1: pack positive_map (C x T, f32 0/1) into WMMA B-fragment layout.
// B matrix is 32(K) x 16(N) f16 per fragment, per ISA layout:
//   lanes 0-15 : N = lane,    halves 0..7 = K 0..7, halves 8..15 = K 16..23 (2 halves per VGPR)
//   lanes 16-31: N = lane-16, halves 0..7 = K 8..15, halves 8..15 = K 24..31
// Fragment id = nt*KSTEPS + ks ; element t = K global = ks*32 + Klocal ; c = N.
// ---------------------------------------------------------------------------
__global__ void __launch_bounds__(128)
pack_pmap_kernel(const float* __restrict__ pmap, _Float16* __restrict__ bfrag) {
    int t = blockIdx.x * blockDim.x + threadIdx.x;   // 0 .. 6*8*32-1
    if (t >= NTILE * KSTEPS * 32) return;
    int lane = t & 31;
    int ks   = (t >> 5) & (KSTEPS - 1);
    int nt   = t >> 8;                               // t / (32*KSTEPS)
    int c     = nt * 16 + (lane & 15);
    int kbase = ks * 32 + ((lane >> 4) << 3);        // +0 or +8
    v16h v;
#pragma unroll
    for (int i = 0; i < 16; ++i) {
        int tt = kbase + i + ((i >> 3) << 3);
        float x = (c < C_) ? pmap[c * T_ + tt] : 0.0f;
        v[i] = (_Float16)x;
    }
    *(v16h*)(bfrag + (size_t)t * 16) = v;
}

// ---------------------------------------------------------------------------
// Kernel 2: batched GEMM with sigmoid fused into the A-fragment conversion.
// One wave per (16-row M tile, batch). A fragment layout (16-bit A 16x32):
//   lanes 0-15 : M = lane,    halves 0..7 = K 0..7,  halves 8..15 = K 16..23
//   lanes 16-31: M = lane-16, halves 0..7 = K 8..15, halves 8..15 = K 24..31
// C/D layout: VGPR r, lanes 0-15 -> (M=r, N=lane); lanes 16-31 -> (M=8+r).
// ---------------------------------------------------------------------------
__global__ void __launch_bounds__(32)
gemm_prob_kernel(const float* __restrict__ logits,
                 const _Float16* __restrict__ bfrag,
                 float* __restrict__ prob) {
    const int mt   = blockIdx.x;          // 0..56
    const int b    = blockIdx.y;          // 0..255
    const int lane = threadIdx.x;         // 0..31
    const int q0   = mt * 16;
    const int m    = lane & 15;
    const int hi   = lane >> 4;           // 0 or 1
    const int q    = q0 + m;
    const int qc   = (q < Q_) ? q : (Q_ - 1);   // clamp: padded rows unused
    const float* rowp = logits + ((size_t)b * Q_ + qc) * T_;

    v8f acc[NTILE] = {};

#pragma unroll
    for (int ks = 0; ks < KSTEPS; ++ks) {
        const int koff = ks * 32 + hi * 8;
        float4 f0 = *(const float4*)(rowp + koff);
        float4 f1 = *(const float4*)(rowp + koff + 4);
        float4 f2 = *(const float4*)(rowp + koff + 16);
        float4 f3 = *(const float4*)(rowp + koff + 20);
        v16h a;
        a[0]  = (_Float16)sigmoidf_(f0.x);
        a[1]  = (_Float16)sigmoidf_(f0.y);
        a[2]  = (_Float16)sigmoidf_(f0.z);
        a[3]  = (_Float16)sigmoidf_(f0.w);
        a[4]  = (_Float16)sigmoidf_(f1.x);
        a[5]  = (_Float16)sigmoidf_(f1.y);
        a[6]  = (_Float16)sigmoidf_(f1.z);
        a[7]  = (_Float16)sigmoidf_(f1.w);
        a[8]  = (_Float16)sigmoidf_(f2.x);
        a[9]  = (_Float16)sigmoidf_(f2.y);
        a[10] = (_Float16)sigmoidf_(f2.z);
        a[11] = (_Float16)sigmoidf_(f2.w);
        a[12] = (_Float16)sigmoidf_(f3.x);
        a[13] = (_Float16)sigmoidf_(f3.y);
        a[14] = (_Float16)sigmoidf_(f3.z);
        a[15] = (_Float16)sigmoidf_(f3.w);
#pragma unroll
        for (int nt = 0; nt < NTILE; ++nt) {
            v16h bb = *(const v16h*)(bfrag + ((size_t)((nt * KSTEPS + ks) * 32 + lane)) * 16);
            acc[nt] = __builtin_amdgcn_wmma_f32_16x16x32_f16(
                false, a, false, bb, (short)0, acc[nt], false, false);
        }
    }

    const int n0 = lane & 15;
    const int mb = hi * 8;
#pragma unroll
    for (int nt = 0; nt < NTILE; ++nt) {
#pragma unroll
        for (int r = 0; r < 8; ++r) {
            int qq = q0 + mb + r;
            int cc = nt * 16 + n0;
            if (qq < Q_ && cc < C_)
                prob[((size_t)b * Q_ + qq) * C_ + cc] = acc[nt][r];
        }
    }
}

// ---------------------------------------------------------------------------
// Kernel 3: exact top-300 per batch (radix select over raw float bits; all
// probs are >= 0 so bit order == value order), deterministic tie-break by
// smallest index, final (value desc, index asc) rank sort, fused box emit.
// ---------------------------------------------------------------------------
#define EQ_CAP 2048

__global__ void __launch_bounds__(256)
topk_emit_kernel(const float* __restrict__ prob,
                 const float* __restrict__ boxes_in,
                 const float* __restrict__ tsizes,
                 float* __restrict__ out) {
    __shared__ unsigned int hist[2048];
    __shared__ unsigned int eqidx[EQ_CAP];
    __shared__ float        sval[K_SEL];
    __shared__ unsigned int sidx[K_SEL];
    __shared__ unsigned int s_prefix, s_dmask, s_krem, s_gcnt, s_ecnt;

    const int b    = blockIdx.x;
    const int tid  = threadIdx.x;
    const int NTOT = Q_ * C_;                       // 81900
    const float* pb = prob + (size_t)b * NTOT;

    if (tid == 0) { s_prefix = 0u; s_dmask = 0u; s_krem = K_SEL; s_gcnt = 0u; s_ecnt = 0u; }
    __syncthreads();

    const int shifts[3] = {21, 10, 0};
    const int nbitsl[3] = {11, 11, 10};

    for (int p = 0; p < 3; ++p) {
        const int shift = shifts[p];
        const unsigned int nb = 1u << nbitsl[p];
        for (int i = tid; i < 2048; i += 256) hist[i] = 0u;
        __syncthreads();
        const unsigned int prefix = s_prefix, dmask = s_dmask;
        for (int i = tid; i < NTOT; i += 256) {
            unsigned int bits = __float_as_uint(pb[i]);
            if ((bits & dmask) == prefix)
                atomicAdd(&hist[(bits >> shift) & (nb - 1u)], 1u);
        }
        __syncthreads();
        if (tid == 0) {
            unsigned int krem = s_krem;
            for (int bin = (int)nb - 1; bin >= 0; --bin) {
                unsigned int c = hist[bin];
                if (c < krem) {
                    krem -= c;
                } else {
                    s_prefix = prefix | ((unsigned int)bin << shift);
                    s_dmask  = dmask | ((nb - 1u) << shift);
                    s_krem   = krem;
                    break;
                }
            }
        }
        __syncthreads();
    }

    const unsigned int thr  = s_prefix;    // exact 32-bit K-th largest value
    const unsigned int krem = s_krem;      // how many equal-to-thr to take

    for (int i = tid; i < NTOT; i += 256) {
        unsigned int bits = __float_as_uint(pb[i]);
        if (bits > thr) {
            unsigned int pos = atomicAdd(&s_gcnt, 1u);
            if (pos < K_SEL) { sval[pos] = __uint_as_float(bits); sidx[pos] = (unsigned int)i; }
        } else if (bits == thr) {
            unsigned int pos = atomicAdd(&s_ecnt, 1u);
            if (pos < EQ_CAP) eqidx[pos] = (unsigned int)i;
        }
    }
    __syncthreads();

    const unsigned int G = (s_gcnt < (unsigned int)K_SEL) ? s_gcnt : (unsigned int)K_SEL;
    const unsigned int E = (s_ecnt < (unsigned int)EQ_CAP) ? s_ecnt : (unsigned int)EQ_CAP;
    const float thrf = __uint_as_float(thr);

    // among equals pick the krem smallest indices (jax tie-break), placed
    // deterministically by ascending-index rank
    for (unsigned int j = tid; j < E; j += 256) {
        const unsigned int me = eqidx[j];
        unsigned int rank = 0;
        for (unsigned int k2 = 0; k2 < E; ++k2) rank += (eqidx[k2] < me) ? 1u : 0u;
        if (rank < krem && (G + rank) < (unsigned int)K_SEL) {
            sval[G + rank] = thrf;
            sidx[G + rank] = me;
        }
    }
    __syncthreads();

    // final deterministic rank sort: (value desc, index asc); all 300 ranks
    // distinct because indices are unique.
    const float img_h = tsizes[b * 2 + 0];
    const float img_w = tsizes[b * 2 + 1];
    float* out_scores = out;
    float* out_labels = out + (size_t)B_ * K_SEL;
    float* out_boxes  = out + (size_t)2 * B_ * K_SEL;

    for (int i = tid; i < K_SEL; i += 256) {
        const float v = sval[i];
        const unsigned int id = sidx[i];
        unsigned int rank = 0;
        for (int j = 0; j < K_SEL; ++j) {
            const float vj = sval[j];
            rank += (vj > v || (vj == v && sidx[j] < id)) ? 1u : 0u;
        }
        const unsigned int qq = id / (unsigned int)C_;
        const unsigned int cc = id - qq * (unsigned int)C_;
        out_scores[(size_t)b * K_SEL + rank] = v;
        out_labels[(size_t)b * K_SEL + rank] = (float)cc;
        const float* bx = boxes_in + ((size_t)b * Q_ + qq) * 4;
        const float cx = bx[0], cy = bx[1], w = bx[2], h = bx[3];
        float* ob = out_boxes + ((size_t)b * K_SEL + rank) * 4;
        ob[0] = (cx - 0.5f * w) * img_w;
        ob[1] = (cy - 0.5f * h) * img_h;
        ob[2] = (cx + 0.5f * w) * img_w;
        ob[3] = (cy + 0.5f * h) * img_h;
    }
}

// ---------------------------------------------------------------------------
extern "C" void kernel_launch(void* const* d_in, const int* in_sizes, int n_in,
                              void* d_out, int out_size, void* d_ws, size_t ws_size,
                              hipStream_t stream) {
    (void)in_sizes; (void)n_in; (void)out_size; (void)ws_size;
    const float* logits = (const float*)d_in[0];   // (256,900,256)
    const float* boxes  = (const float*)d_in[1];   // (256,900,4)
    const float* pmap   = (const float*)d_in[2];   // (91,256)
    const float* tsz    = (const float*)d_in[3];   // (256,2)

    _Float16* bfrag = (_Float16*)((char*)d_ws + WS_BFRAG_OFF);
    float*    probw = (float*)((char*)d_ws + WS_PROB_OFF);
    float*    out   = (float*)d_out;

    pack_pmap_kernel<<<dim3(12), dim3(128), 0, stream>>>(pmap, bfrag);
    gemm_prob_kernel<<<dim3(MTILE, B_), dim3(32), 0, stream>>>(logits, bfrag, probw);
    topk_emit_kernel<<<dim3(B_), dim3(256), 0, stream>>>(probw, boxes, tsz, out);
}
